// MultiHeadAttention_39986145526235
// MI455X (gfx1250) — matrix operand order repaired and checked by reading, not compiled, not run on hardware
//
#include <hip/hip_runtime.h>
#include <stdint.h>

// ---------------- problem dims (fixed by reference) ----------------
#define BB    4
#define SS    2048
#define DD    1024
#define HH    16
#define DHD   64
#define RR    12
#define BH    (BB * HH)      // 64
#define MROWS (BB * SS)      // 8192 rows for the projection GEMMs
#define CTX_M (BB * 384)     // 1536 rows for the final GEMM

// ---------------- WMMA types (CDNA5, wave32) ----------------
typedef __attribute__((ext_vector_type(16))) __bf16 bf16x16;
typedef __attribute__((ext_vector_type(8)))  float  f32x8;

union Frag {
    bf16x16        v;
    uint4          q[2];
    unsigned short h[16];
};

__device__ __forceinline__ unsigned short f2bf(float f) {
    unsigned int u = __float_as_uint(f);
    unsigned int r = (u + 0x7FFFu + ((u >> 16) & 1u)) >> 16;  // round-to-nearest-even
    return (unsigned short)r;
}

__device__ __forceinline__ f32x8 wmma_bf16(const Frag& a, const Frag& b, f32x8 c) {
    // D = A(16x32 bf16) x B(32x16 bf16) + C(16x16 f32)
    return __builtin_amdgcn_wmma_f32_16x16x32_bf16(false, a.v, false, b.v,
                                                   (short)0, c, false, false);
}

// ---- CDNA5 async global->LDS copy (ASYNCcnt tracked), GV addressing ----
// LDS generic addresses are {SHARED_BASE[63:32], lds_offset[31:0]} (ISA 10.2
// aperture rule), so truncating the generic pointer yields the LDS byte offset
// the instruction's VDST operand wants.
__device__ __forceinline__ void async_copy_b128(void* lds_ptr, const void* gptr) {
    const unsigned           ldsOff = (unsigned)(size_t)lds_ptr;
    const unsigned long long gaddr  = (unsigned long long)(size_t)gptr;
    asm volatile("global_load_async_to_lds_b128 %0, %1, off"
                 :: "v"(ldsOff), "v"(gaddr) : "memory");
}
__device__ __forceinline__ void wait_async0() {
    asm volatile("s_wait_asynccnt 0x0" ::: "memory");
}

// ============================================================================
// Kernel 0: fp32 -> bf16 elementwise conversion (vectorized x4).
// One pass over Q/K/V/weights so the GEMM hot loops carry no convert VALU work
// and can stage tiles with pure async copies.
// ============================================================================
__global__ __launch_bounds__(256) void cvt_bf16_kernel(
    const float* __restrict__ in, unsigned short* __restrict__ out, int n4)
{
    const int i = blockIdx.x * 256 + threadIdx.x;
    if (i < n4) {
        const float4 f = ((const float4*)in)[i];
        ushort4 o;
        o.x = f2bf(f.x); o.y = f2bf(f.y); o.z = f2bf(f.z); o.w = f2bf(f.w);
        ((ushort4*)out)[i] = o;
    }
}

// ============================================================================
// Kernel 1: bf16 GEMM  C = A @ W^T   (A:[M,1024], W:[1024,1024], both bf16)
// Block 256 thr = 8 waves; block tile 128x128, K-step 32 (= one WMMA K).
// Double-buffered LDS staged with global_load_async_to_lds_b128:
//   issue async(k+1) -> compute 8 WMMAs on tile k -> s_wait_asynccnt 0 -> barrier.
// MODE 0: bf16 out scattered to [B,H,S,DH] (projection).
// MODE 1: fp32 out + bias (final linear, pre-LayerNorm).
// ============================================================================
template <int MODE>
__global__ __launch_bounds__(256) void gemm_bf16_kernel(
    const unsigned short* __restrict__ A,    // bf16 [M, 1024]
    const unsigned short* __restrict__ Bw,   // bf16 [1024, 1024] row-major weights
    const float* __restrict__ bias,          // MODE 1 only
    unsigned short* __restrict__ obf,        // MODE 0 out
    float* __restrict__ of32)                // MODE 1 out
{
    __shared__ __align__(16) unsigned short ldsA[2][128][40];  // +8 pad, 80B rows
    __shared__ __align__(16) unsigned short ldsB[2][128][40];

    const int tid  = threadIdx.x;
    const int lane = tid & 31;
    const int wave = tid >> 5;
    const int m0 = blockIdx.x * 128;
    const int n0 = blockIdx.y * 128;
    const int wm = (wave & 1) * 64;
    const int wn = (wave >> 1) * 32;

    // Each tile is 128x32 halves = 8KB = 512 16B-chunks; 2 chunks per thread.
    const int id0 = tid * 2, id1 = id0 + 1;
    const int r0 = id0 >> 2, g0 = (id0 & 3) * 8;
    const int r1 = id1 >> 2, g1 = (id1 & 3) * 8;

    f32x8 acc[4][2] = {};

    // prologue: stage k=0 into buffer 0
    async_copy_b128(&ldsA[0][r0][g0], A  + (size_t)(m0 + r0) * DD + g0);
    async_copy_b128(&ldsA[0][r1][g1], A  + (size_t)(m0 + r1) * DD + g1);
    async_copy_b128(&ldsB[0][r0][g0], Bw + (size_t)(n0 + r0) * DD + g0);
    async_copy_b128(&ldsB[0][r1][g1], Bw + (size_t)(n0 + r1) * DD + g1);
    wait_async0();
    __syncthreads();

    for (int k0 = 0; k0 < DD; k0 += 32) {
        const int buf = (k0 >> 5) & 1;
        if (k0 + 32 < DD) {   // uniform: EXEC stays full around WMMAs
            const int kn = k0 + 32;
            async_copy_b128(&ldsA[buf ^ 1][r0][g0], A  + (size_t)(m0 + r0) * DD + kn + g0);
            async_copy_b128(&ldsA[buf ^ 1][r1][g1], A  + (size_t)(m0 + r1) * DD + kn + g1);
            async_copy_b128(&ldsB[buf ^ 1][r0][g0], Bw + (size_t)(n0 + r0) * DD + kn + g0);
            async_copy_b128(&ldsB[buf ^ 1][r1][g1], Bw + (size_t)(n0 + r1) * DD + kn + g1);
        }

        // B frag (32x16): N=lane%16; elems 0..7: K 0..7 / 8..15, elems 8..15: K 16..23 / 24..31
        Frag bf[2];
        #pragma unroll
        for (int nt = 0; nt < 2; ++nt) {
            const int n  = wn + nt * 16 + (lane & 15);
            const int kb = (lane < 16) ? 0 : 8;
            bf[nt].q[0] = *(const uint4*)&ldsB[buf][n][kb];
            bf[nt].q[1] = *(const uint4*)&ldsB[buf][n][kb + 16];
        }
        // A frag (16x32): M=lane%16; VGPR0..3: K 0..7 / 8..15, VGPR4..7: K 16..23 / 24..31
        #pragma unroll
        for (int mt = 0; mt < 4; ++mt) {
            Frag af;
            const int m  = wm + mt * 16 + (lane & 15);
            const int kb = (lane < 16) ? 0 : 8;
            af.q[0] = *(const uint4*)&ldsA[buf][m][kb];
            af.q[1] = *(const uint4*)&ldsA[buf][m][16 + kb];
            #pragma unroll
            for (int nt = 0; nt < 2; ++nt)
                acc[mt][nt] = wmma_bf16(af, bf[nt], acc[mt][nt]);
        }

        wait_async0();      // my async writes for tile k+1 have landed
        __syncthreads();    // everyone done reading buf & done writing buf^1
    }

    // C layout: element r -> M = r + (lane<16?0:8), N = lane%16
    #pragma unroll
    for (int mt = 0; mt < 4; ++mt) {
        #pragma unroll
        for (int nt = 0; nt < 2; ++nt) {
            const int gn = n0 + wn + nt * 16 + (lane & 15);
            if (MODE == 0) {
                const int h  = gn >> 6;
                const int dh = gn & 63;
                #pragma unroll
                for (int r = 0; r < 8; ++r) {
                    const int gm = m0 + wm + mt * 16 + r + ((lane < 16) ? 0 : 8);
                    const int b  = gm >> 11;
                    const int s  = gm & 2047;
                    obf[(((size_t)(b * HH + h) * SS + s) * DHD) + dh] = f2bf(acc[mt][nt][r]);
                }
            } else {
                const float bl = bias[gn];
                #pragma unroll
                for (int r = 0; r < 8; ++r) {
                    const int gm = m0 + wm + mt * 16 + r + ((lane < 16) ? 0 : 8);
                    of32[(size_t)gm * DD + gn] = acc[mt][nt][r] + bl;
                }
            }
        }
    }
}

// ============================================================================
// Kernel 2: s1 = q @ U^T per (b,h).  q:[S,64] bf16, U:[12,64] f32 (zero-pad to 16).
// s1 stored TRANSPOSED as bf16 [BH,16,S] so phase 3 gets contiguous K.
// ============================================================================
__global__ __launch_bounds__(256) void s1_kernel(
    const unsigned short* __restrict__ qp,   // bf16 [BH,S,DH]
    const float* __restrict__ U,             // f32 [B,H,R,DH]
    unsigned short* __restrict__ s1)         // bf16 [BH,16,S]
{
    __shared__ __align__(16) unsigned short ldsU[16][64];   // rows 128B
    const int bh  = blockIdx.y;
    const int s0  = blockIdx.x * 128;
    const int tid = threadIdx.x;

    for (int i = tid; i < 16 * 64; i += 256) {
        const int r = i >> 6, d = i & 63;
        ldsU[r][d] = (r < RR) ? f2bf(U[((size_t)bh * RR + r) * DHD + d]) : (unsigned short)0;
    }
    __syncthreads();

    const int lane = tid & 31, wave = tid >> 5;
    const int mrow = s0 + wave * 16 + (lane & 15);
    const unsigned short* qrow = qp + ((size_t)bh * SS + mrow) * DHD;

    f32x8 acc = {};
    #pragma unroll
    for (int k0 = 0; k0 < DHD; k0 += 32) {
        Frag a, b;
        const int ak = k0 + ((lane < 16) ? 0 : 8);
        a.q[0] = *(const uint4*)(qrow + ak);
        a.q[1] = *(const uint4*)(qrow + ak + 16);
        const int bk = k0 + ((lane < 16) ? 0 : 8);   // B[k=dh][n=r] = U[r][dh]
        b.q[0] = *(const uint4*)&ldsU[lane & 15][bk];
        b.q[1] = *(const uint4*)&ldsU[lane & 15][bk + 16];
        acc = wmma_bf16(a, b, acc);
    }

    const int rcol = lane & 15;
    #pragma unroll
    for (int r = 0; r < 8; ++r) {
        const int s = s0 + wave * 16 + r + ((lane < 16) ? 0 : 8);
        s1[((size_t)bh * 16 + rcol) * SS + s] = f2bf(acc[r]);
    }
}

// ============================================================================
// Kernel 3: s2[r,d] = sum_s s1[s,r]*k[s,d]  (K = S = 2048), fused softmax over d.
// One block per (b,h); 8 waves split K (256 each), LDS reduce, softmax.
// ============================================================================
__global__ __launch_bounds__(256) void s2_softmax_kernel(
    const unsigned short* __restrict__ s1,   // bf16 [BH,16,S]
    const unsigned short* __restrict__ kp,   // bf16 [BH,S,DH]
    unsigned short* __restrict__ attn)       // bf16 [BH,16,DH]
{
    __shared__ float red[8][16][64];
    __shared__ float s2s[16][64];
    const int bh   = blockIdx.x;
    const int tid  = threadIdx.x;
    const int lane = tid & 31, wave = tid >> 5;
    const unsigned short* s1p = s1 + (size_t)bh * 16 * SS;
    const unsigned short* kpp = kp + (size_t)bh * SS * DHD;

    f32x8 acc[4] = {};
    const int r   = lane & 15;
    const int akb = (lane < 16) ? 0 : 8;
    for (int kk = 0; kk < 256; kk += 32) {
        const int sBase = wave * 256 + kk;
        Frag a;   // A[m=r][k=s] = s1p[r][s] (contiguous along s)
        a.q[0] = *(const uint4*)(s1p + (size_t)r * SS + sBase + akb);
        a.q[1] = *(const uint4*)(s1p + (size_t)r * SS + sBase + 16 + akb);
        const int ks = sBase + ((lane < 16) ? 0 : 8);
        #pragma unroll
        for (int nt = 0; nt < 4; ++nt) {
            Frag b;   // B[k=s][n=d] = kpp[s][d] : strided gather
            const int d = nt * 16 + (lane & 15);
            #pragma unroll
            for (int e = 0; e < 16; ++e)
                b.h[e] = kpp[(size_t)(ks + ((e < 8) ? e : e + 8)) * DHD + d];
            acc[nt] = wmma_bf16(a, b, acc[nt]);
        }
    }
    #pragma unroll
    for (int nt = 0; nt < 4; ++nt)
        #pragma unroll
        for (int rr = 0; rr < 8; ++rr)
            red[wave][rr + ((lane < 16) ? 0 : 8)][nt * 16 + (lane & 15)] = acc[nt][rr];
    __syncthreads();

    for (int i = tid; i < 16 * 64; i += 256) {
        const int m = i >> 6, n = i & 63;
        float s = 0.f;
        #pragma unroll
        for (int w = 0; w < 8; ++w) s += red[w][m][n];
        s2s[m][n] = s;
    }
    __syncthreads();

    if (tid < 16) {                       // softmax over dh=64 (after all WMMAs)
        float mx = -3.0e38f;
        for (int n = 0; n < 64; ++n) mx = fmaxf(mx, s2s[tid][n]);
        float sum = 0.f;
        for (int n = 0; n < 64; ++n) sum += __expf(s2s[tid][n] - mx);
        const float inv = 1.f / sum;
        for (int n = 0; n < 64; ++n)
            attn[((size_t)bh * 16 + tid) * DHD + n] = f2bf(__expf(s2s[tid][n] - mx) * inv);
    }
}

// ============================================================================
// Kernel 4: ctx[r,s] = sum_d attn[r,d]*v[s,d], scattered straight into the
// reshaped [B,384,1024] bf16 layout (rows r>=12 never consumed downstream).
// ============================================================================
__global__ __launch_bounds__(256) void ctx_kernel(
    const unsigned short* __restrict__ attn, // bf16 [BH,16,DH]
    const unsigned short* __restrict__ vp,   // bf16 [BH,S,DH]
    unsigned short* __restrict__ ctx)        // bf16 [B,384,1024]
{
    const int bh = blockIdx.y;
    const int b = bh >> 4, h = bh & 15;
    const int tid = threadIdx.x;
    const int lane = tid & 31, wave = tid >> 5;
    const unsigned short* ap  = attn + (size_t)bh * 16 * DHD;
    const unsigned short* vpp = vp + (size_t)bh * SS * DHD;
    const int nbase = blockIdx.x * 256 + wave * 32;

    f32x8 acc[2] = {};
    #pragma unroll
    for (int k0 = 0; k0 < DHD; k0 += 32) {
        Frag a;   // A[m=r][k=d] = attn[r][d] (contiguous)
        const int rr = lane & 15;
        const int kb = k0 + ((lane < 16) ? 0 : 8);
        a.q[0] = *(const uint4*)(ap + (size_t)rr * DHD + kb);
        a.q[1] = *(const uint4*)(ap + (size_t)rr * DHD + kb + 16);
        const int d0 = k0 + ((lane < 16) ? 0 : 8);
        #pragma unroll
        for (int nt = 0; nt < 2; ++nt) {
            Frag bb;  // B[k=d][n=s] = v[s][d] (contiguous along d)
            const int s = nbase + nt * 16 + (lane & 15);
            bb.q[0] = *(const uint4*)(vpp + (size_t)s * DHD + d0);
            bb.q[1] = *(const uint4*)(vpp + (size_t)s * DHD + d0 + 16);
            acc[nt] = wmma_bf16(a, bb, acc[nt]);
        }
    }
    #pragma unroll
    for (int nt = 0; nt < 2; ++nt) {
        const int s = nbase + nt * 16 + (lane & 15);
        #pragma unroll
        for (int r = 0; r < 8; ++r) {
            const int rr = r + ((lane < 16) ? 0 : 8);
            if (rr < RR) {   // divergent store, all WMMAs already done
                const int mrow = rr * 32 + h * 2 + (s >> 10);
                ctx[((size_t)b * 384 + mrow) * DD + (s & 1023)] = f2bf(acc[nt][r]);
            }
        }
    }
}

// ============================================================================
// Kernel 5: LayerNorm over last dim (1024), eps=1e-6, population variance.
// ============================================================================
__global__ __launch_bounds__(256) void layernorm_kernel(
    const float* __restrict__ X,      // [CTX_M, DD]
    const float* __restrict__ gamma,
    const float* __restrict__ beta,
    float* __restrict__ out)
{
    __shared__ float rs[256], rq[256];
    const int tid = threadIdx.x;
    const float* row = X + (size_t)blockIdx.x * DD;
    float lsum = 0.f, lsq = 0.f;
    for (int i = tid; i < DD; i += 256) {
        const float v = row[i];
        lsum += v; lsq += v * v;
    }
    rs[tid] = lsum; rq[tid] = lsq;
    __syncthreads();
    for (int off = 128; off > 0; off >>= 1) {
        if (tid < off) { rs[tid] += rs[tid + off]; rq[tid] += rq[tid + off]; }
        __syncthreads();
    }
    const float mu  = rs[0] * (1.f / DD);
    const float var = rq[0] * (1.f / DD) - mu * mu;
    const float inv = rsqrtf(var + 1e-6f);
    for (int i = tid; i < DD; i += 256)
        out[(size_t)blockIdx.x * DD + i] = (row[i] - mu) * inv * gamma[i] + beta[i];
}

// ============================================================================
// Host launcher
// ============================================================================
extern "C" void kernel_launch(void* const* d_in, const int* in_sizes, int n_in,
                              void* d_out, int out_size, void* d_ws, size_t ws_size,
                              hipStream_t stream) {
    (void)in_sizes; (void)n_in; (void)out_size; (void)ws_size;

    const float* Q     = (const float*)d_in[0];
    const float* K     = (const float*)d_in[1];
    const float* V     = (const float*)d_in[2];
    const float* U     = (const float*)d_in[3];
    const float* W_Q   = (const float*)d_in[4];
    const float* W_K   = (const float*)d_in[5];
    const float* W_V   = (const float*)d_in[6];
    const float* W_lin = (const float*)d_in[7];
    const float* b_lin = (const float*)d_in[8];
    const float* gamma = (const float*)d_in[9];
    const float* beta  = (const float*)d_in[10];
    float* out = (float*)d_out;

    // ---- workspace layout (halves unless noted), total ~104 MB ----
    const size_t ACT = (size_t)MROWS * DD;     // 8,388,608 halves = 16MB
    const size_t WEI = (size_t)DD * DD;        // 1,048,576 halves = 2MB
    unsigned short* Qb  = (unsigned short*)d_ws;   // bf16 copies of inputs
    unsigned short* Kb  = Qb  + ACT;
    unsigned short* Vb  = Kb  + ACT;
    unsigned short* Wqb = Vb  + ACT;
    unsigned short* Wkb = Wqb + WEI;
    unsigned short* Wvb = Wkb + WEI;
    unsigned short* Wlb = Wvb + WEI;
    unsigned short* qp  = Wlb + WEI;           // projection outputs [BH,S,DH]
    unsigned short* kp  = qp  + ACT;
    unsigned short* vp  = kp  + ACT;
    // phase-2 scratch ALIASED into Qb (dead after the projections): 13.2MB <= 16MB
    unsigned short* s1w  = Qb;                              // [BH,16,S]   4MB
    unsigned short* attn = s1w  + (size_t)BH * 16 * SS;     // [BH,16,64]  128KB
    unsigned short* ctxw = attn + (size_t)BH * 16 * DHD;    // [B,384,1024] 3MB
    float* outpre = (float*)(ctxw + (size_t)CTX_M * DD);    // [CTX_M,DD] f32 6MB

    const dim3 blk(256);

    // Phase 0: one-shot fp32 -> bf16 conversions (L2-resident working set)
    cvt_bf16_kernel<<<dim3((unsigned)(ACT / 4 / 256)), blk, 0, stream>>>(Q, Qb, (int)(ACT / 4));
    cvt_bf16_kernel<<<dim3((unsigned)(ACT / 4 / 256)), blk, 0, stream>>>(K, Kb, (int)(ACT / 4));
    cvt_bf16_kernel<<<dim3((unsigned)(ACT / 4 / 256)), blk, 0, stream>>>(V, Vb, (int)(ACT / 4));
    cvt_bf16_kernel<<<dim3((unsigned)(WEI / 4 / 256)), blk, 0, stream>>>(W_Q,   Wqb, (int)(WEI / 4));
    cvt_bf16_kernel<<<dim3((unsigned)(WEI / 4 / 256)), blk, 0, stream>>>(W_K,   Wkb, (int)(WEI / 4));
    cvt_bf16_kernel<<<dim3((unsigned)(WEI / 4 / 256)), blk, 0, stream>>>(W_V,   Wvb, (int)(WEI / 4));
    cvt_bf16_kernel<<<dim3((unsigned)(WEI / 4 / 256)), blk, 0, stream>>>(W_lin, Wlb, (int)(WEI / 4));

    // Phase 1: projections (bulk of FLOPs), async-staged bf16 WMMA GEMMs
    gemm_bf16_kernel<0><<<dim3(MROWS / 128, DD / 128), blk, 0, stream>>>(Qb, Wqb, nullptr, qp, nullptr);
    gemm_bf16_kernel<0><<<dim3(MROWS / 128, DD / 128), blk, 0, stream>>>(Kb, Wkb, nullptr, kp, nullptr);
    gemm_bf16_kernel<0><<<dim3(MROWS / 128, DD / 128), blk, 0, stream>>>(Vb, Wvb, nullptr, vp, nullptr);

    // Phase 2: random projection s1 = q @ U^T (stored transposed)
    s1_kernel<<<dim3(SS / 128, BH), blk, 0, stream>>>(qp, U, s1w);

    // Phase 3: s2 = s1^T @ k, fused softmax over dh
    s2_softmax_kernel<<<dim3(BH), blk, 0, stream>>>(s1w, kp, attn);

    // Phase 4: ctx = attn @ v^T, scattered into reshaped layout
    ctx_kernel<<<dim3(SS / 256, BH), blk, 0, stream>>>(attn, vp, ctxw);

    // Phase 5: final linear + bias (fp32 out)
    gemm_bf16_kernel<1><<<dim3(CTX_M / 128, DD / 128), blk, 0, stream>>>(ctxw, Wlb, b_lin, nullptr, outpre);

    // Phase 6: LayerNorm
    layernorm_kernel<<<dim3(CTX_M), blk, 0, stream>>>(outpre, gamma, beta, out);
}
